// SelfAttention_36146444764042
// MI455X (gfx1250) — compile-verified, matrix operand order checked by reading, not run on hardware
//
#include <hip/hip_runtime.h>
#include <cstdint>
#include <cstddef>

// ---------------------------------------------------------------------------
// Self-attention (PointNet-style) for MI455X / gfx1250, wave32 + WMMA f16.
//   q,k,v = BN(Conv1d(x))  -> S = q^T k -> softmax -> out = A v
// B=8, C=512, N=2048.  All contractions on v_wmma_f32_16x16x32_f16.
// v3: projection GEMM stages the workgroup-shared x tile through LDS with
//     gfx1250 async copies (global_load_async_to_lds_b128 / s_wait_asynccnt),
//     double-buffered per K-step: 8x less VMEM traffic, ASYNCcnt-decoupled.
// Workspace requirement: ~68 MB (xT, Wh, shift, qT, kT, vN).
// ---------------------------------------------------------------------------

typedef __attribute__((ext_vector_type(16))) _Float16 v16h;
typedef __attribute__((ext_vector_type(8)))  _Float16 v8h;
typedef __attribute__((ext_vector_type(8)))  float    v8f;

constexpr int B_ = 8, C_ = 512, N_ = 2048;
constexpr float LOG2E = 1.4426950408889634f;
constexpr float BN_EPS = 1e-5f;

static __device__ __forceinline__ v16h ld16h(const _Float16* p) {
  return *(const v16h*)p;
}
static __device__ __forceinline__ v8f wmma16(v16h a, v16h b, v8f c) {
  // D = A(16x32) * B(32x16) + C(16x16), f16 in / f32 acc
  return __builtin_amdgcn_wmma_f32_16x16x32_f16(false, a, false, b,
                                                (short)0, c, false, false);
}
// Async 16B copy global -> LDS (ASYNCcnt domain).
static __device__ __forceinline__ void async_cp16(uint32_t lds_addr,
                                                  const void* gptr) {
  asm volatile("global_load_async_to_lds_b128 %0, %1, off"
               :: "v"(lds_addr), "v"((uint64_t)(uintptr_t)gptr)
               : "memory");
}
static __device__ __forceinline__ void wait_async0() {
  asm volatile("s_wait_asynccnt 0x0" ::: "memory");
}

struct Proj { const float *W, *g, *bb, *m, *v; };

// ---------------------------------------------------------------------------
// Stage 0a: x (B,C,N) f32 -> xT (B,N,C) f16 via LDS tile transpose.
// ---------------------------------------------------------------------------
__global__ __launch_bounds__(256) void k_transpose_cast(const float* __restrict__ x,
                                                        _Float16* __restrict__ xT) {
  __shared__ _Float16 tile[32][33];
  const int b  = blockIdx.z;
  const int n0 = blockIdx.x * 32;
  const int c0 = blockIdx.y * 32;
  const int tx = threadIdx.x, ty = threadIdx.y;   // block (32, 8)
#pragma unroll
  for (int r = 0; r < 4; ++r) {
    int c = c0 + ty + 8 * r;
    tile[ty + 8 * r][tx] = (_Float16)x[((size_t)b * C_ + c) * N_ + n0 + tx];
  }
  __syncthreads();
#pragma unroll
  for (int r = 0; r < 4; ++r) {
    int n = n0 + ty + 8 * r;
    xT[((size_t)b * N_ + n) * C_ + c0 + tx] = tile[tx][ty + 8 * r];
  }
}

// ---------------------------------------------------------------------------
// Stage 0b: fold BN scale into W rows (f32->f16), precompute shift.
// ---------------------------------------------------------------------------
__global__ __launch_bounds__(256) void k_prep_weights(Proj pq, Proj pk, Proj pv,
                                                      _Float16* __restrict__ Wh,
                                                      float* __restrict__ shift) {
  const int p = blockIdx.y;
  const Proj pr = (p == 0) ? pq : ((p == 1) ? pk : pv);
  const int idx = blockIdx.x * 256 + threadIdx.x;   // [0, C*C)
  const int o = idx / C_;
  const float sc = pr.g[o] * rsqrtf(pr.v[o] + BN_EPS);
  Wh[(size_t)p * C_ * C_ + idx] = (_Float16)(pr.W[idx] * sc);
  if (idx < C_) {
    float sc2 = pr.g[idx] * rsqrtf(pr.v[idx] + BN_EPS);
    shift[p * C_ + idx] = pr.bb[idx] - pr.m[idx] * sc2;
  }
}

// ---------------------------------------------------------------------------
// Stage 1: projection GEMM  y[o,n] = sum_c Wh[o,c] xT[n,c] + shift[o]
// Workgroup = 128o x 64n; the shared 64-row x tile is streamed through LDS
// with double-buffered async copies (one 4KB chunk per K-step).
// Each wave: 16o x 64n, K=512 in 16 WMMA steps; A (W rows) pipelined in regs.
// LDS rows padded to 40 halfs (20 DW, gcd(20,64)=4 -> conflict-free b128 reads).
// ---------------------------------------------------------------------------
__global__ __launch_bounds__(256) void k_proj_gemm(const _Float16* __restrict__ Wh,
                                                   const float* __restrict__ shift,
                                                   const _Float16* __restrict__ xT,
                                                   _Float16* __restrict__ qT,
                                                   _Float16* __restrict__ kT,
                                                   _Float16* __restrict__ vN) {
  __shared__ alignas(16) _Float16 lds_x[2][64][40];   // 10 KB, double buffer

  const int tid = threadIdx.x;
  const int wave = tid >> 5, lane = tid & 31;
  const int l16 = lane & 15, lh = lane >> 4;
  const int b = blockIdx.z % B_;
  const int p = blockIdx.z / B_;
  const int o0 = blockIdx.y * 128 + wave * 16;
  const int n0 = blockIdx.x * 64;

  // one 16B async packet per thread per K-chunk: 64 rows x 4 packets
  const int srow = tid >> 2;            // 0..63
  const int sq   = (tid & 3) * 8;       // half offset within 32-half chunk
  const _Float16* xsrc = xT + ((size_t)b * N_ + n0 + srow) * C_ + sq;

  const _Float16* Wrow = Wh + (size_t)p * C_ * C_ + (size_t)(o0 + l16) * C_ + 16 * lh;

  v8f acc[4];
#pragma unroll
  for (int t = 0; t < 4; ++t) acc[t] = (v8f)0.0f;

  // prologue: chunk 0 into buffer 0, first A fragment
  async_cp16((uint32_t)(uintptr_t)&lds_x[0][srow][sq], xsrc);
  v16h a_cur = ld16h(Wrow);
  wait_async0();
  __syncthreads();

#pragma unroll
  for (int ks = 0; ks < 16; ++ks) {
    const int buf = ks & 1;
    if (ks < 15)
      async_cp16((uint32_t)(uintptr_t)&lds_x[buf ^ 1][srow][sq],
                 xsrc + (ks + 1) * 32);
    v16h a_nxt = ld16h(Wrow + ((ks < 15) ? (ks + 1) : 15) * 32);
#pragma unroll
    for (int t = 0; t < 4; ++t) {
      const _Float16* row = &lds_x[buf][16 * t + l16][16 * lh];
      v8h lo = *(const v8h*)row;
      v8h hi = *(const v8h*)(row + 8);
      v16h bm = __builtin_shufflevector(lo, hi, 0, 1, 2, 3, 4, 5, 6, 7,
                                        8, 9, 10, 11, 12, 13, 14, 15);
      acc[t] = wmma16(a_cur, bm, acc[t]);
    }
    a_cur = a_nxt;
    wait_async0();       // next chunk landed in LDS
    __syncthreads();     // all waves done reading current buffer
  }

  float sh[8];
#pragma unroll
  for (int r = 0; r < 8; ++r) sh[r] = shift[p * C_ + o0 + r + 8 * lh];

  if (p < 2) {
    _Float16* dst = (p == 0) ? qT : kT;
#pragma unroll
    for (int t = 0; t < 4; ++t) {
      v8h hv;
#pragma unroll
      for (int r = 0; r < 8; ++r) hv[r] = (_Float16)(acc[t][r] + sh[r]);
      *(v8h*)(dst + ((size_t)b * N_ + n0 + 16 * t + l16) * C_ + o0 + 8 * lh) = hv;
    }
  } else {
#pragma unroll
    for (int t = 0; t < 4; ++t)
#pragma unroll
      for (int r = 0; r < 8; ++r)
        vN[((size_t)b * C_ + o0 + r + 8 * lh) * N_ + n0 + 16 * t + l16] =
            (_Float16)(acc[t][r] + sh[r]);
  }
}

// ---------------------------------------------------------------------------
// Stage 2: flash attention.  One workgroup = (b, 16-query tile), 8 waves.
// j-blocks of 128 (16 keys/wave).  O: 16i x 512c, 64c per wave.
// S uses two interleaved accumulators; k/v operand loads double-buffered.
// (k/v slices are wave-disjoint: direct register loads, no LDS hop.)
// ---------------------------------------------------------------------------
__global__ __launch_bounds__(256) void k_flash_attn(const _Float16* __restrict__ qT,
                                                    const _Float16* __restrict__ kT,
                                                    const _Float16* __restrict__ vN,
                                                    float* __restrict__ out) {
  __shared__ float lds_red[8][16];
  __shared__ float lds_m[16], lds_l[16], lds_alpha[16];
  __shared__ alignas(32) _Float16 lds_P[16][128];

  const int tid = threadIdx.x;
  const int wave = tid >> 5, lane = tid & 31;
  const int l16 = lane & 15, lh = lane >> 4;
  const int b = blockIdx.y;
  const int i0 = blockIdx.x * 16;

  // q A-fragments: 16 K-steps, kept register resident (128 VGPRs).
  v16h qf[16];
  {
    const _Float16* qb = qT + ((size_t)b * N_ + i0) * C_ + (size_t)l16 * C_;
#pragma unroll
    for (int ks = 0; ks < 16; ++ks) qf[ks] = ld16h(qb + ks * 32 + 16 * lh);
  }

  v8f oacc[4];
#pragma unroll
  for (int t = 0; t < 4; ++t) oacc[t] = (v8f)0.0f;
  if (tid < 16) { lds_m[tid] = -INFINITY; lds_l[tid] = 0.0f; }

  for (int jb = 0; jb < N_ / 128; ++jb) {
    const int j0 = jb * 128 + wave * 16;

    // ---- S = q^T k : dual accumulator, double-buffered k fragments ----
    const _Float16* kb = kT + ((size_t)b * N_ + j0) * C_ + (size_t)l16 * C_ + 16 * lh;
    if (jb + 1 < N_ / 128)
      __builtin_prefetch(kb + (size_t)128 * C_, 0, 1);

    v8f s0 = (v8f)0.0f, s1 = (v8f)0.0f;
    v16h kf_cur = ld16h(kb);
#pragma unroll
    for (int ks = 0; ks < 16; ++ks) {
      v16h kf_nxt = ld16h(kb + ((ks < 15) ? (ks + 1) : 15) * 32);
      if (ks & 1) s1 = wmma16(qf[ks], kf_cur, s1);
      else        s0 = wmma16(qf[ks], kf_cur, s0);
      kf_cur = kf_nxt;
    }
    v8f s = (s0 + s1) * LOG2E;   // exp2 domain

    // ---- per-wave row max over the 16 j columns (lanes of each half) ----
    float red[8];
#pragma unroll
    for (int r = 0; r < 8; ++r) {
      float m = s[r];
#pragma unroll
      for (int off = 8; off >= 1; off >>= 1)
        m = fmaxf(m, __shfl_xor(m, off, 32));
      red[r] = m;
    }
    if (l16 == 0) {
#pragma unroll
      for (int r = 0; r < 8; ++r) lds_red[wave][8 * lh + r] = red[r];
    }
    __syncthreads();                                  // (A)

    // ---- workgroup online-softmax state update ----
    if (tid < 16) {
      float mb = lds_red[0][tid];
#pragma unroll
      for (int w = 1; w < 8; ++w) mb = fmaxf(mb, lds_red[w][tid]);
      float mo = lds_m[tid];
      float mn = fmaxf(mo, mb);
      float al = exp2f(mo - mn);
      lds_alpha[tid] = al;
      lds_m[tid] = mn;
      lds_l[tid] *= al;
    }
    __syncthreads();                                  // (B)

    // ---- P = exp2(S - m), row sums, stage P in LDS, rescale O ----
    float mrow[8], ar[8], p[8];
#pragma unroll
    for (int r = 0; r < 8; ++r) {
      mrow[r] = lds_m[r + 8 * lh];
      ar[r]   = lds_alpha[r + 8 * lh];
    }
#pragma unroll
    for (int r = 0; r < 8; ++r) {
      p[r] = exp2f(s[r] - mrow[r]);
      lds_P[r + 8 * lh][wave * 16 + l16] = (_Float16)p[r];
      float sm = p[r];
#pragma unroll
      for (int off = 8; off >= 1; off >>= 1)
        sm += __shfl_xor(sm, off, 32);
      red[r] = sm;
    }
    if (l16 == 0) {
#pragma unroll
      for (int r = 0; r < 8; ++r) lds_red[wave][8 * lh + r] = red[r];
    }
#pragma unroll
    for (int t = 0; t < 4; ++t)
#pragma unroll
      for (int r = 0; r < 8; ++r) oacc[t][r] *= ar[r];
    __syncthreads();                                  // (C)

    if (tid < 16) {
      float ssum = lds_red[0][tid];
#pragma unroll
      for (int w = 1; w < 8; ++w) ssum += lds_red[w][tid];
      lds_l[tid] += ssum;
    }

    // ---- O += P(16x128) * V^T(128x64 slice), pipelined over 16 WMMAs ----
    const _Float16* vb = vN + (size_t)b * C_ * N_ + jb * 128 + 16 * lh
                       + (size_t)(wave * 64 + l16) * N_;
    // flattened idx = ks*4 + ct ; vf address = vb + ct*16*N_ + ks*32
    v16h vf_cur = ld16h(vb);
    v16h pf_cur = *(const v16h*)(&lds_P[l16][16 * lh]);
#pragma unroll
    for (int idx = 0; idx < 16; ++idx) {
      const int ks = idx >> 2, ct = idx & 3;
      const int ni = (idx < 15) ? idx + 1 : 15;
      v16h vf_nxt = ld16h(vb + (size_t)(ni & 3) * 16 * N_ + (ni >> 2) * 32);
      oacc[ct] = wmma16(pf_cur, vf_cur, oacc[ct]);
      if ((ni >> 2) != ks)
        pf_cur = *(const v16h*)(&lds_P[l16][(ni >> 2) * 32 + 16 * lh]);
      vf_cur = vf_nxt;
    }
    __syncthreads();                                  // (D) end of j-block
  }

  // ---- normalize by l and store out (B,C,N) f32, 32B packed per lane ----
  float linv[8];
#pragma unroll
  for (int r = 0; r < 8; ++r) linv[r] = 1.0f / lds_l[r + 8 * lh];
#pragma unroll
  for (int ct = 0; ct < 4; ++ct) {
    v8f ov;
#pragma unroll
    for (int r = 0; r < 8; ++r) ov[r] = oacc[ct][r] * linv[r];
    float* dst = out + ((size_t)b * C_ + wave * 64 + ct * 16 + l16) * N_
                     + i0 + 8 * lh;
    *(v8f*)dst = ov;
  }
}

// ---------------------------------------------------------------------------
extern "C" void kernel_launch(void* const* d_in, const int* in_sizes, int n_in,
                              void* d_out, int out_size, void* d_ws, size_t ws_size,
                              hipStream_t stream) {
  (void)in_sizes; (void)n_in; (void)out_size; (void)ws_size;
  const float* x = (const float*)d_in[0];
  Proj pq{(const float*)d_in[1],  (const float*)d_in[2],  (const float*)d_in[3],
          (const float*)d_in[4],  (const float*)d_in[5]};
  Proj pk{(const float*)d_in[6],  (const float*)d_in[7],  (const float*)d_in[8],
          (const float*)d_in[9],  (const float*)d_in[10]};
  Proj pv{(const float*)d_in[11], (const float*)d_in[12], (const float*)d_in[13],
          (const float*)d_in[14], (const float*)d_in[15]};
  float* out = (float*)d_out;

  char* ws = (char*)d_ws;
  size_t off = 0;
  auto alloc = [&](size_t bytes) {
    void* p = ws + off;
    off = (off + bytes + 255) & ~(size_t)255;
    return p;
  };
  _Float16* xT   = (_Float16*)alloc((size_t)B_ * N_ * C_ * 2);
  _Float16* Wh   = (_Float16*)alloc((size_t)3 * C_ * C_ * 2);
  float*    shft = (float*)   alloc((size_t)3 * C_ * 4);
  _Float16* qT   = (_Float16*)alloc((size_t)B_ * N_ * C_ * 2);
  _Float16* kT   = (_Float16*)alloc((size_t)B_ * N_ * C_ * 2);
  _Float16* vN   = (_Float16*)alloc((size_t)B_ * C_ * N_ * 2);

  k_transpose_cast<<<dim3(N_ / 32, C_ / 32, B_), dim3(32, 8), 0, stream>>>(x, xT);
  k_prep_weights<<<dim3(C_ * C_ / 256, 3), 256, 0, stream>>>(pq, pk, pv, Wh, shft);
  k_proj_gemm<<<dim3(N_ / 64, C_ / 128, 3 * B_), 256, 0, stream>>>(Wh, shft, xT,
                                                                   qT, kT, vN);
  k_flash_attn<<<dim3(N_ / 16, B_), 256, 0, stream>>>(qT, kT, vN, out);
}
